// TrtPostRunner_3470333575522
// MI455X (gfx1250) — matrix operand order repaired and checked by reading, not compiled, not run on hardware
//
#include <hip/hip_runtime.h>
#include <hip/hip_bf16.h>

typedef __attribute__((ext_vector_type(16))) _Float16 v16h;
typedef __attribute__((ext_vector_type(8)))  _Float16 v8h;
typedef __attribute__((ext_vector_type(8)))  float    v8f;

__device__ __forceinline__ float actf(float v, int act) {
  if (act == 1) return fmaxf(v, 0.f);
  if (act == 2) return 1.f / (1.f + __expf(-v));
  if (act == 3) return tanhf(v);
  return v;
}
__device__ __forceinline__ int clampi(int v, int lo, int hi) {
  return v < lo ? lo : (v > hi ? hi : v);
}

// ---------------------------------------------------------------------------
// Weight packer: f32 [Cout,Cin,ks,ks] -> f16 [CoutPad, ks*ks*CinPad],
// K ordered (ky,kx,ci) so every 32-wide K-step sits in one (ky,kx) slab.
// ---------------------------------------------------------------------------
__global__ void k_pack_w(const float* __restrict__ wt, _Float16* __restrict__ wh,
                         int Cout, int Cin, int CinPad, int ks, int CoutPad)
{
  int KKp = ks * ks * CinPad;
  int tid = blockIdx.x * blockDim.x + threadIdx.x;
  if (tid >= CoutPad * KKp) return;
  int m = tid / KKp, k = tid % KKp;
  int slab = k / CinPad, ci = k % CinPad;
  int ky = slab / ks, kx = slab % ks;
  float v = 0.f;
  if (m < Cout && ci < Cin) v = wt[((m * Cin + ci) * ks + ky) * ks + kx];
  wh[tid] = (_Float16)v;
}

// f32 channel-major [C,HW] -> f16 pixel-major [HW,Cpad] (zero padded)
__global__ void k_to_f16_pm(const float* __restrict__ src, _Float16* __restrict__ dst,
                            int C, int Cpad, int HWl)
{
  int tid = blockIdx.x * blockDim.x + threadIdx.x;
  if (tid >= HWl * Cpad) return;
  int p = tid / Cpad, c = tid % Cpad;
  dst[tid] = (_Float16)((c < C) ? src[c * HWl + p] : 0.f);
}

// ---------------------------------------------------------------------------
// WMMA implicit-GEMM conv2d. Input: f16 pixel-major [HW, CinPad] (CinPad%32==0).
// Weights: f16 [CoutPad, KS*KS*CinPad]. One wave computes 16 out-ch x 32 pixels
// (two 16x16 tiles sharing the A fragment). All fragment loads are aligned
// b128 vector loads; one boundary predicate per K-step per tile.
// Output: f32 channel-major (outh==null) or f16 pixel-major [HW,ocs] at ocoff.
// ---------------------------------------------------------------------------
template <int KS>
__global__ __launch_bounds__(32)
void k_wmma_conv(const _Float16* __restrict__ in, int CinPad,
                 const _Float16* __restrict__ wh, const float* __restrict__ bias,
                 float* __restrict__ outf, _Float16* __restrict__ outh,
                 int ocs, int ocoff, int Cout, int H, int W, int act)
{
  const int lane  = threadIdx.x;
  const int grpW  = W >> 5;                       // groups of 32 pixels per row
  const int h     = blockIdx.x / grpW;
  const int w0    = (blockIdx.x % grpW) << 5;
  const int mBase = blockIdx.y << 4;
  const int KK    = KS * KS * CinPad;
  const int m     = mBase + (lane & 15);
  const int n     = lane & 15;
  const int koffA = (lane >> 4) << 3;
  const int koffB = (lane >> 4) << 3;
  const _Float16* wrow = wh + (size_t)m * KK;
  const int HWl = H * W;

  v8f acc0 = {}, acc1 = {};
  for (int k0 = 0; k0 < KK; k0 += 32) {
    // ---- A fragment: two contiguous 8-half runs per lane ----
    const _Float16* ap = wrow + k0 + koffA;
    v8h a0 = *(const v8h*)ap;
    v8h a1 = *(const v8h*)(ap + 16);
    __builtin_prefetch(ap + 32, 0, 1);
    v16h a;
#pragma unroll
    for (int i = 0; i < 8; ++i) { a[i] = a0[i]; a[i + 8] = a1[i]; }
    // ---- B fragments: whole step shares one (ky,kx) tap ----
    const int slab = k0 / CinPad;
    const int ci0  = k0 - slab * CinPad;
    const int ky   = slab / KS, kx = slab - ky * KS;
    const int ih   = h + ky - (KS >> 1);
    const int ihok = (ih >= 0) & (ih < H);
    const int ihc  = clampi(ih, 0, H - 1);
#pragma unroll
    for (int t = 0; t < 2; ++t) {
      int w  = w0 + (t << 4) + n;
      int iw = w + kx - (KS >> 1);
      int ok = ihok & (iw >= 0) & (iw < W);
      int iwc = clampi(iw, 0, W - 1);
      const _Float16* bp = in + (size_t)(ihc * W + iwc) * CinPad + ci0 + koffB;
      v8h b0 = *(const v8h*)bp;
      v8h b1 = *(const v8h*)(bp + 16);
      _Float16 s = (_Float16)(ok ? 1.f : 0.f);
      v16h b;
#pragma unroll
      for (int i = 0; i < 8; ++i) { b[i] = b0[i] * s; b[i + 8] = b1[i] * s; }
      if (t == 0)
        acc0 = __builtin_amdgcn_wmma_f32_16x16x32_f16(false, a, false, b,
                                                      (short)0, acc0, false, false);
      else
        acc1 = __builtin_amdgcn_wmma_f32_16x16x32_f16(false, a, false, b,
                                                      (short)0, acc1, false, false);
    }
  }
#pragma unroll
  for (int r = 0; r < 8; ++r) {
    int mo = mBase + r + ((lane >> 4) << 3);
    if (mo < Cout) {
      float bsv = bias[mo];
      int wa = w0 + n, wb = w0 + 16 + n;
      float va = actf(acc0[r] + bsv, act);
      float vb = actf(acc1[r] + bsv, act);
      if (outh) {
        outh[(size_t)(h * W + wa) * ocs + ocoff + mo] = (_Float16)va;
        outh[(size_t)(h * W + wb) * ocs + ocoff + mo] = (_Float16)vb;
      } else {
        outf[(size_t)mo * HWl + h * W + wa] = va;
        outf[(size_t)mo * HWl + h * W + wb] = vb;
      }
    }
  }
}

// ---------------------------------------------------------------------------
// WMMA correlation from pixel-major f16 features:
// corr[h, wl, wr] = sum_c L[h*W+wl, c] * R[h*W+wr, c] * scale   (C % 32 == 0)
// ---------------------------------------------------------------------------
__global__ __launch_bounds__(32)
void k_corr(const _Float16* __restrict__ L, const _Float16* __restrict__ R,
            float* __restrict__ corr, int C, int H, int W, float scale)
{
  const int lane  = threadIdx.x;
  const int tiles = W >> 4;
  const int tm    = (blockIdx.x / tiles) << 4;
  const int tn    = (blockIdx.x % tiles) << 4;
  const int h     = blockIdx.y;
  const int wl    = tm + (lane & 15);
  const int wr    = tn + (lane & 15);
  const int koffA = (lane >> 4) << 3;
  const int koffB = (lane >> 4) << 3;
  const _Float16* lp = L + (size_t)(h * W + wl) * C;
  const _Float16* rp = R + (size_t)(h * W + wr) * C;
  v8f acc = {};
  for (int k0 = 0; k0 < C; k0 += 32) {
    v8h a0 = *(const v8h*)(lp + k0 + koffA);
    v8h a1 = *(const v8h*)(lp + k0 + koffA + 16);
    v8h b0 = *(const v8h*)(rp + k0 + koffB);
    v8h b1 = *(const v8h*)(rp + k0 + koffB + 16);
    v16h a, b;
#pragma unroll
    for (int i = 0; i < 8; ++i) {
      a[i] = a0[i]; a[i + 8] = a1[i];
      b[i] = b0[i]; b[i + 8] = b1[i];
    }
    acc = __builtin_amdgcn_wmma_f32_16x16x32_f16(false, a, false, b,
                                                 (short)0, acc, false, false);
  }
#pragma unroll
  for (int r = 0; r < 8; ++r) {
    int row = tm + r + ((lane >> 4) << 3);
    corr[(h * W + row) * W + wr] = acc[r] * scale;
  }
}

// ---------------------------------------------------------------------------
// stem 3D conv: builds concat([gwc(8), lvol(12), rvol(12)]) on the fly, relu.
// ---------------------------------------------------------------------------
__global__ void k_stem(const float* __restrict__ gwc, const float* __restrict__ pl,
                       const float* __restrict__ pr, const float* __restrict__ wt,
                       const float* __restrict__ bias, float* __restrict__ out,
                       int Dd, int H, int W)
{
  int HWl = H * W;
  int tid = blockIdx.x * blockDim.x + threadIdx.x;
  if (tid >= 8 * Dd * HWl) return;
  int p  = tid % HWl;
  int d  = (tid / HWl) % Dd;
  int co = tid / (HWl * Dd);
  int h = p / W, w = p % W;
  float s = bias[co];
  for (int ic = 0; ic < 32; ++ic) {
    const float* wp = wt + (co * 32 + ic) * 27;
    for (int kd = 0; kd < 3; ++kd) {
      int dd = d + kd - 1;
      if (dd < 0 || dd >= Dd) continue;
      for (int kh = 0; kh < 3; ++kh) {
        int hh = h + kh - 1;
        if (hh < 0 || hh >= H) continue;
        for (int kw = 0; kw < 3; ++kw) {
          int ww = w + kw - 1;
          if (ww < 0 || ww >= W) continue;
          float v;
          if (ic < 8)        v = gwc[(ic * Dd + dd) * HWl + hh * W + ww];
          else if (ww < dd)  v = 0.f;
          else if (ic < 20)  v = pl[(ic - 8) * HWl + hh * W + ww];
          else               v = pr[(ic - 20) * HWl + hh * W + (ww - dd)];
          s += v * wp[kd * 9 + kh * 3 + kw];
        }
      }
    }
  }
  out[tid] = fmaxf(s, 0.f);
}

// generic 3x3x3 conv, Cin=8, optional per-(c,h,w) attention scale on the input
__global__ void k_conv3d_c8(const float* __restrict__ vin, const float* __restrict__ att,
                            const float* __restrict__ wt, const float* __restrict__ bias,
                            float* __restrict__ out, int Cout, int Dd, int H, int W, int relu)
{
  int HWl = H * W;
  int tid = blockIdx.x * blockDim.x + threadIdx.x;
  if (tid >= Cout * Dd * HWl) return;
  int p  = tid % HWl;
  int d  = (tid / HWl) % Dd;
  int co = tid / (HWl * Dd);
  int h = p / W, w = p % W;
  float s = bias[co];
  for (int ic = 0; ic < 8; ++ic) {
    const float* wp = wt + (co * 8 + ic) * 27;
    for (int kd = 0; kd < 3; ++kd) {
      int dd = d + kd - 1;
      if (dd < 0 || dd >= Dd) continue;
      for (int kh = 0; kh < 3; ++kh) {
        int hh = h + kh - 1;
        if (hh < 0 || hh >= H) continue;
        for (int kw = 0; kw < 3; ++kw) {
          int ww = w + kw - 1;
          if (ww < 0 || ww >= W) continue;
          float v = vin[(ic * Dd + dd) * HWl + hh * W + ww];
          if (att) v *= att[ic * HWl + hh * W + ww];
          s += v * wp[kd * 9 + kh * 3 + kw];
        }
      }
    }
  }
  out[tid] = relu ? fmaxf(s, 0.f) : s;
}

__global__ void k_softmax_disp(const float* __restrict__ logits, float* __restrict__ disp,
                               int Dd, int HWl)
{
  int p = blockIdx.x * blockDim.x + threadIdx.x;
  if (p >= HWl) return;
  float m = -1e30f;
  for (int d = 0; d < Dd; ++d) m = fmaxf(m, logits[d * HWl + p]);
  float s = 0.f, e = 0.f;
  for (int d = 0; d < Dd; ++d) {
    float ex = __expf(logits[d * HWl + p] - m);
    s += ex; e += ex * (float)d;
  }
  disp[p] = e / s;
}

// direct conv3x3 (small Cout): sam / med / head.
// Output f32 channel-major, or f16 pixel-major [HW,ocs] at ocoff if outh given.
__global__ void k_conv3x3_direct(const float* __restrict__ in, const float* __restrict__ wt,
                                 const float* __restrict__ bias, float* __restrict__ outf,
                                 _Float16* __restrict__ outh, int ocs, int ocoff,
                                 int Cin, int Cout, int H, int W, int act)
{
  int HWl = H * W;
  int tid = blockIdx.x * blockDim.x + threadIdx.x;
  if (tid >= Cout * HWl) return;
  int p = tid % HWl, co = tid / HWl;
  int h = p / W, w = p % W;
  float s = bias[co];
  for (int ci = 0; ci < Cin; ++ci) {
    const float* wp = wt + (co * Cin + ci) * 9;
    for (int ky = 0; ky < 3; ++ky) {
      int hh = h + ky - 1;
      if (hh < 0 || hh >= H) continue;
      for (int kx = 0; kx < 3; ++kx) {
        int ww = w + kx - 1;
        if (ww < 0 || ww >= W) continue;
        s += in[ci * HWl + hh * W + ww] * wp[ky * 3 + kx];
      }
    }
  }
  float v = actf(s, act);
  if (outh) outh[(size_t)p * ocs + ocoff + co] = (_Float16)v;
  else      outf[tid] = v;
}

__global__ void k_chan_mean(const float* __restrict__ x, float* __restrict__ mean, int HWl)
{
  __shared__ float sm[256];
  int c = blockIdx.x;
  float s = 0.f;
  for (int i = threadIdx.x; i < HWl; i += 256) s += x[c * HWl + i];
  sm[threadIdx.x] = s;
  __syncthreads();
  for (int st = 128; st > 0; st >>= 1) {
    if (threadIdx.x < st) sm[threadIdx.x] += sm[threadIdx.x + st];
    __syncthreads();
  }
  if (threadIdx.x == 0) mean[c] = sm[0] / (float)HWl;
}

__global__ void k_cam(const float* __restrict__ mean, const float* __restrict__ w,
                      const float* __restrict__ b, float* __restrict__ sig, int C)
{
  int co = threadIdx.x;
  if (co >= C) return;
  float s = b[co];
  for (int ci = 0; ci < C; ++ci) s += w[co * C + ci] * mean[ci];
  sig[co] = 1.f / (1.f + __expf(-s));
}

__global__ void k_scale_ch(float* __restrict__ x, const float* __restrict__ sig,
                           int C, int HWl)
{
  int tid = blockIdx.x * blockDim.x + threadIdx.x;
  if (tid >= C * HWl) return;
  x[tid] *= sig[tid / HWl];
}

__global__ void k_pool_d(const float* __restrict__ vin, float* __restrict__ vout,
                         int C, int D2, int HWl)
{
  int tid = blockIdx.x * blockDim.x + threadIdx.x;
  if (tid >= C * D2 * HWl) return;
  int p = tid % HWl;
  int d = (tid / HWl) % D2;
  int c = tid / (HWl * D2);
  const float* src = vin + (c * (2 * D2) + 2 * d) * HWl + p;
  vout[tid] = 0.5f * (src[0] + src[HWl]);
}

__global__ void k_pool_last(const float* __restrict__ cin, float* __restrict__ cout,
                            int rows, int L2)
{
  int tid = blockIdx.x * blockDim.x + threadIdx.x;
  if (tid >= rows * L2) return;
  int i = tid % L2, row = tid / L2;
  cout[tid] = 0.5f * (cin[row * 2 * L2 + 2 * i] + cin[row * 2 * L2 + 2 * i + 1]);
}

// geo feature sampling -> f16 pixel-major [HW, 192] (channels 162..191 zeroed)
__global__ void k_geo(const float* __restrict__ gv0, const float* __restrict__ gv1,
                      const float* __restrict__ c0, const float* __restrict__ c1,
                      const float* __restrict__ disp, _Float16* __restrict__ geo,
                      int H, int W)
{
  int HWl = H * W;
  int tid = blockIdx.x * blockDim.x + threadIdx.x;
  if (tid >= 2 * 9 * HWl) return;
  int p = tid % HWl;
  int k = (tid / HWl) % 9;
  int l = tid / (9 * HWl);
  int w = p % W;
  float dv  = disp[p];
  float inv = l ? 0.5f : 1.f;
  int base  = l ? 81 : 0;
  const float* gv = l ? gv1 : gv0;
  int Dl = l ? 24 : 48;
  float pos = dv * inv + (float)(k - 4);
  float fl = floorf(pos);
  float f  = pos - fl;
  int i0 = clampi((int)fl, 0, Dl - 1);
  int i1 = (i0 + 1 < Dl) ? i0 + 1 : Dl - 1;
  _Float16* gp = geo + (size_t)p * 192;
  for (int c = 0; c < 8; ++c) {
    float v0 = gv[(c * Dl + i0) * HWl + p];
    float v1 = gv[(c * Dl + i1) * HWl + p];
    gp[base + c * 9 + k] = (_Float16)(v0 * (1.f - f) + v1 * f);
  }
  const float* cc = l ? c1 : c0;
  int Ll = l ? (W >> 1) : W;
  float cpos = ((float)w - dv) * inv + (float)(k - 4);
  float cfl = floorf(cpos);
  float cf  = cpos - cfl;
  int j0 = clampi((int)cfl, 0, Ll - 1);
  int j1 = (j0 + 1 < Ll) ? j0 + 1 : Ll - 1;
  gp[base + 72 + k] = (_Float16)(cc[p * Ll + j0] * (1.f - cf) + cc[p * Ll + j1] * cf);
  if (l == 0 && k == 0)
    for (int c = 162; c < 192; ++c) gp[c] = (_Float16)0.f;
}

// elementwise helpers
__global__ void k_copy(float* __restrict__ d, const float* __restrict__ s, int n)
{ int i = blockIdx.x * blockDim.x + threadIdx.x; if (i < n) d[i] = s[i]; }

__global__ void k_add(float* __restrict__ d, const float* __restrict__ s, int n)
{ int i = blockIdx.x * blockDim.x + threadIdx.x; if (i < n) d[i] += s[i]; }

__global__ void k_build_x(float* __restrict__ x, const float* __restrict__ inp,
                          const float* __restrict__ motion, const float* __restrict__ satt,
                          int HWl)
{
  int tid = blockIdx.x * blockDim.x + threadIdx.x;
  if (tid >= 256 * HWl) return;
  int c = tid / HWl, p = tid % HWl;
  x[tid] = (c < 128) ? inp[tid] : motion[(c - 128) * HWl + p] * satt[p];
}

// hx (f16 pixel-major [HW,384]) = concat(net, x)
__global__ void k_build_hx(_Float16* __restrict__ hx, const float* __restrict__ net,
                           const float* __restrict__ x, int HWl)
{
  int tid = blockIdx.x * blockDim.x + threadIdx.x;
  if (tid >= 384 * HWl) return;
  int p = tid / 384, c = tid % 384;
  float v = (c < 128) ? net[c * HWl + p] : x[(c - 128) * HWl + p];
  hx[tid] = (_Float16)v;
}

// hx[:,0:128] = r * net   (in place, pixel-major f16)
__global__ void k_rnx(_Float16* __restrict__ hx, const float* __restrict__ r,
                      const float* __restrict__ net, int HWl)
{
  int tid = blockIdx.x * blockDim.x + threadIdx.x;
  if (tid >= 128 * HWl) return;
  int c = tid / HWl, p = tid % HWl;
  hx[(size_t)p * 384 + c] = (_Float16)(r[tid] * net[tid]);
}

__global__ void k_net_update(float* __restrict__ net, const float* __restrict__ z,
                             const float* __restrict__ q, int HWl)
{
  int tid = blockIdx.x * blockDim.x + threadIdx.x;
  if (tid >= 128 * HWl) return;
  float zz = z[tid];
  net[tid] = (1.f - zz) * net[tid] + zz * q[tid];
}

// cat2x (f16 pixel-major [4HW, 64]) = concat(repeat2(maskf), stem_2x)
__global__ void k_cat2x(const float* __restrict__ maskf, const float* __restrict__ stem2x,
                        _Float16* __restrict__ cat2x, int H, int W)
{
  int W2 = 2 * W, HW2 = 4 * H * W, HWl = H * W;
  int tid = blockIdx.x * blockDim.x + threadIdx.x;
  if (tid >= 64 * HW2) return;
  int p2 = tid / 64, c = tid % 64;
  int y = p2 / W2, x = p2 % W2;
  float v = (c < 32) ? maskf[c * HWl + (y >> 1) * W + (x >> 1)]
                     : stem2x[(c - 32) * HW2 + p2];
  cat2x[tid] = (_Float16)v;
}

// Fused: transposed-conv(4x4,s2) -> softmax(9) -> convex combination of the
// unfolded 3x3 disparity neighborhood, repeated x4.
__global__ void k_upsample(const float* __restrict__ xspx, const float* __restrict__ wg,
                           const float* __restrict__ bg, const float* __restrict__ disp,
                           float* __restrict__ out, int H, int W)
{
  int H2 = 2 * H, W2 = 2 * W, W4 = 4 * W;
  int tid = blockIdx.x * blockDim.x + threadIdx.x;
  if (tid >= 16 * H * W) return;
  int y = tid / W4, x = tid % W4;
  float lg[9];
#pragma unroll
  for (int o = 0; o < 9; ++o) lg[o] = bg[o];
  for (int ky = 0; ky < 4; ++ky) {
    int iy2 = y + 1 - ky;
    if (iy2 < 0 || (iy2 & 1)) continue;
    int iy = iy2 >> 1;
    if (iy >= H2) continue;
    for (int kx = 0; kx < 4; ++kx) {
      int ix2 = x + 1 - kx;
      if (ix2 < 0 || (ix2 & 1)) continue;
      int ix = ix2 >> 1;
      if (ix >= W2) continue;
      for (int i = 0; i < 32; ++i) {
        float v = xspx[(i * H2 + iy) * W2 + ix];
#pragma unroll
        for (int o = 0; o < 9; ++o)
          lg[o] += v * wg[(o * 32 + i) * 16 + ky * 4 + kx];
      }
    }
  }
  float mx = lg[0];
#pragma unroll
  for (int o = 1; o < 9; ++o) mx = fmaxf(mx, lg[o]);
  float s = 0.f;
#pragma unroll
  for (int o = 0; o < 9; ++o) { lg[o] = __expf(lg[o] - mx); s += lg[o]; }
  float invs = 1.f / s;
  int h = y >> 2, w = x >> 2;
  float acc = 0.f;
  for (int i = 0; i < 3; ++i)
    for (int j = 0; j < 3; ++j) {
      int hh = h + i - 1, ww = w + j - 1;
      float dv = (hh >= 0 && hh < H && ww >= 0 && ww < W) ? disp[hh * W + ww] * 4.f : 0.f;
      acc += lg[i * 3 + j] * invs * dv;
    }
  out[tid] = acc;
}

// ---------------------------------------------------------------------------
extern "C" void kernel_launch(void* const* d_in, const int* in_sizes, int n_in,
                              void* d_out, int out_size, void* d_ws, size_t ws_size,
                              hipStream_t stream)
{
  (void)in_sizes; (void)n_in; (void)out_size; (void)ws_size;
  const int H = 96, W = 160, Dd = 48, HWl = H * W;

  const float* f4     = (const float*)d_in[0];
  const float* fr4    = (const float*)d_in[4];
  const float* stem2x = (const float*)d_in[5];
  const float* gwc    = (const float*)d_in[6];
  const float* proj_w = (const float*)d_in[7];  const float* proj_b = (const float*)d_in[8];
  const float* stem_w = (const float*)d_in[9];  const float* stem_b = (const float*)d_in[10];
  const float* att_w  = (const float*)d_in[11]; const float* att_b  = (const float*)d_in[12];
  const float* agg_w  = (const float*)d_in[13]; const float* agg_b  = (const float*)d_in[14];
  const float* cls_w  = (const float*)d_in[15]; const float* cls_b  = (const float*)d_in[16];
  const float* cnn_w  = (const float*)d_in[17]; const float* cnn_b  = (const float*)d_in[18];
  const float* cni_w  = (const float*)d_in[19]; const float* cni_b  = (const float*)d_in[20];
  const float* cam_w  = (const float*)d_in[21]; const float* cam_b  = (const float*)d_in[22];
  const float* sam_w  = (const float*)d_in[23]; const float* sam_b  = (const float*)d_in[24];
  const float* meg_w  = (const float*)d_in[25]; const float* meg_b  = (const float*)d_in[26];
  const float* med_w  = (const float*)d_in[27]; const float* med_b  = (const float*)d_in[28];
  const float* meo_w  = (const float*)d_in[29]; const float* meo_b  = (const float*)d_in[30];
  const float* gz_w   = (const float*)d_in[31]; const float* gz_b   = (const float*)d_in[32];
  const float* gr_w   = (const float*)d_in[33]; const float* gr_b   = (const float*)d_in[34];
  const float* gq_w   = (const float*)d_in[35]; const float* gq_b   = (const float*)d_in[36];
  const float* head_w = (const float*)d_in[37]; const float* head_b = (const float*)d_in[38];
  const float* mask_w = (const float*)d_in[39]; const float* mask_b = (const float*)d_in[40];
  const float* spx2_w = (const float*)d_in[41]; const float* spx2_b = (const float*)d_in[42];
  const float* spxg_w = (const float*)d_in[43]; const float* spxg_b = (const float*)d_in[44];

  float* ws = (float*)d_ws;
  size_t off = 0;
  auto alloc  = [&](size_t nf) { off = (off + 7) & ~(size_t)7; float* p = ws + off; off += nf; return p; };
  auto allocH = [&](size_t nh) { return (_Float16*)alloc((nh + 1) / 2); };
  const size_t DHW = (size_t)Dd * HWl;

  // f32 buffers
  float* pl     = alloc(12 * (size_t)HWl);
  float* pr     = alloc(12 * (size_t)HWl);
  float* stemv  = alloc(8 * DHW);
  float* aggv   = alloc(8 * DHW);
  float* gv1    = alloc(8 * 24 * (size_t)HWl);
  float* att    = alloc(8 * (size_t)HWl);
  float* logits = alloc(DHW);
  float* disp   = alloc(HWl);
  float* ddisp  = alloc(HWl);
  float* net    = alloc(128 * (size_t)HWl);
  float* inp    = alloc(128 * (size_t)HWl);
  float* satt   = alloc(HWl);
  float* camv   = alloc(256);
  float* motion = alloc(128 * (size_t)HWl);
  float* xb     = alloc(256 * (size_t)HWl);
  float* zb     = alloc(128 * (size_t)HWl);
  float* rb     = alloc(128 * (size_t)HWl);
  float* maskf  = alloc(32 * (size_t)HWl);
  // f16 pixel-major activations
  _Float16* f4h   = allocH((size_t)HWl * 96);
  _Float16* fr4h  = allocH((size_t)HWl * 96);
  _Float16* geoh  = allocH((size_t)HWl * 192);
  _Float16* hxh   = allocH((size_t)HWl * 384);
  _Float16* mgmdh = allocH((size_t)HWl * 160);
  // packed f16 weights
  _Float16* whp  = allocH(16 * 96);           // proj
  _Float16* wha  = allocH(16 * 96);           // att
  _Float16* whn  = allocH(128 * 864);         // cnet net
  _Float16* whi  = allocH(128 * 864);         // cnet inp
  _Float16* whmg = allocH(96 * 192);          // meg (Cin 162 -> 192)
  _Float16* whmo = allocH(128 * 1440);        // meo (Cout 127 -> 128)
  _Float16* whz  = allocH(128 * 3456);
  _Float16* whr  = allocH(128 * 3456);
  _Float16* whq  = allocH(128 * 3456);
  _Float16* whmk = allocH(32 * 1152);         // mask
  _Float16* whs2 = allocH(32 * 576);          // spx2
  // aliased scratch (lifetimes disjoint on the stream):
  float*    corr0  = stemv;                     // stem volume dead after agg
  float*    corr1  = stemv + (size_t)HWl * W;
  _Float16* cat2xh = hxh;                       // dead after GRU iterations
  _Float16* neth   = geoh;                      // dead after GRU iterations
  float*    xspx   = xb;                        // dead after GRU iterations

  auto cdiv = [](int a, int b) { return (a + b - 1) / b; };
  dim3 wv(32);

  // --- weight packing + f16 transposed feature copies ---
  k_pack_w<<<cdiv(16 * 96, 256), 256, 0, stream>>>(proj_w, whp, 12, 96, 96, 1, 16);
  k_pack_w<<<cdiv(16 * 96, 256), 256, 0, stream>>>(att_w, wha, 8, 96, 96, 1, 16);
  k_pack_w<<<cdiv(128 * 864, 256), 256, 0, stream>>>(cnn_w, whn, 128, 96, 96, 3, 128);
  k_pack_w<<<cdiv(128 * 864, 256), 256, 0, stream>>>(cni_w, whi, 128, 96, 96, 3, 128);
  k_pack_w<<<cdiv(96 * 192, 256), 256, 0, stream>>>(meg_w, whmg, 96, 162, 192, 1, 96);
  k_pack_w<<<cdiv(128 * 1440, 256), 256, 0, stream>>>(meo_w, whmo, 127, 160, 160, 3, 128);
  k_pack_w<<<cdiv(128 * 3456, 256), 256, 0, stream>>>(gz_w, whz, 128, 384, 384, 3, 128);
  k_pack_w<<<cdiv(128 * 3456, 256), 256, 0, stream>>>(gr_w, whr, 128, 384, 384, 3, 128);
  k_pack_w<<<cdiv(128 * 3456, 256), 256, 0, stream>>>(gq_w, whq, 128, 384, 384, 3, 128);
  k_pack_w<<<cdiv(32 * 1152, 256), 256, 0, stream>>>(mask_w, whmk, 32, 128, 128, 3, 32);
  k_pack_w<<<cdiv(32 * 576, 256), 256, 0, stream>>>(spx2_w, whs2, 32, 64, 64, 3, 32);
  k_to_f16_pm<<<cdiv(HWl * 96, 256), 256, 0, stream>>>(f4, f4h, 96, 96, HWl);
  k_to_f16_pm<<<cdiv(HWl * 96, 256), 256, 0, stream>>>(fr4, fr4h, 96, 96, HWl);

  // --- cost volume construction + aggregation ---
  k_wmma_conv<1><<<dim3(HWl / 32, 1), wv, 0, stream>>>(f4h, 96, whp, proj_b, pl, nullptr, 0, 0, 12, H, W, 0);
  k_wmma_conv<1><<<dim3(HWl / 32, 1), wv, 0, stream>>>(fr4h, 96, whp, proj_b, pr, nullptr, 0, 0, 12, H, W, 0);
  k_stem<<<cdiv(8 * Dd * HWl, 256), 256, 0, stream>>>(gwc, pl, pr, stem_w, stem_b, stemv, Dd, H, W);
  k_wmma_conv<1><<<dim3(HWl / 32, 1), wv, 0, stream>>>(f4h, 96, wha, att_b, att, nullptr, 0, 0, 8, H, W, 2);
  k_conv3d_c8<<<cdiv(8 * Dd * HWl, 256), 256, 0, stream>>>(stemv, att, agg_w, agg_b, aggv, 8, Dd, H, W, 1);
  k_conv3d_c8<<<cdiv(Dd * HWl, 256), 256, 0, stream>>>(aggv, nullptr, cls_w, cls_b, logits, 1, Dd, H, W, 0);
  k_softmax_disp<<<cdiv(HWl, 256), 256, 0, stream>>>(logits, disp, Dd, HWl);

  // --- context network + attentions ---
  k_wmma_conv<3><<<dim3(HWl / 32, 8), wv, 0, stream>>>(f4h, 96, whn, cnn_b, net, nullptr, 0, 0, 128, H, W, 3);
  k_wmma_conv<3><<<dim3(HWl / 32, 8), wv, 0, stream>>>(f4h, 96, whi, cni_b, inp, nullptr, 0, 0, 128, H, W, 1);
  k_chan_mean<<<128, 256, 0, stream>>>(inp, camv, HWl);
  k_cam<<<1, 128, 0, stream>>>(camv, cam_w, cam_b, camv + 128, 128);
  k_scale_ch<<<cdiv(128 * HWl, 256), 256, 0, stream>>>(inp, camv + 128, 128, HWl);
  k_conv3x3_direct<<<cdiv(HWl, 256), 256, 0, stream>>>(inp, sam_w, sam_b, satt, nullptr, 0, 0, 128, 1, H, W, 2);

  // --- geo encoding pyramids (corr0/corr1 alias dead stem volume) ---
  k_corr<<<dim3((W / 16) * (W / 16), H), wv, 0, stream>>>(f4h, fr4h, corr0, 96, H, W, 0.1020620726f);
  k_pool_last<<<cdiv(HWl * (W / 2), 256), 256, 0, stream>>>(corr0, corr1, HWl, W / 2);
  k_pool_d<<<cdiv(8 * 24 * HWl, 256), 256, 0, stream>>>(aggv, gv1, 8, 24, HWl);

  // --- GRU refinement iterations ---
  for (int it = 0; it < 4; ++it) {
    k_geo<<<cdiv(2 * 9 * HWl, 256), 256, 0, stream>>>(aggv, gv1, corr0, corr1, disp, geoh, H, W);
    k_wmma_conv<1><<<dim3(HWl / 32, 6), wv, 0, stream>>>(geoh, 192, whmg, meg_b, nullptr, mgmdh, 160, 0, 96, H, W, 1);
    k_conv3x3_direct<<<cdiv(64 * HWl, 256), 256, 0, stream>>>(disp, med_w, med_b, nullptr, mgmdh, 160, 96, 1, 64, H, W, 1);
    k_wmma_conv<3><<<dim3(HWl / 32, 8), wv, 0, stream>>>(mgmdh, 160, whmo, meo_b, motion, nullptr, 0, 0, 127, H, W, 1);
    k_copy<<<cdiv(HWl, 256), 256, 0, stream>>>(motion + 127 * (size_t)HWl, disp, HWl);
    k_build_x<<<cdiv(256 * HWl, 256), 256, 0, stream>>>(xb, inp, motion, satt, HWl);
    k_build_hx<<<cdiv(384 * HWl, 256), 256, 0, stream>>>(hxh, net, xb, HWl);
    k_wmma_conv<3><<<dim3(HWl / 32, 8), wv, 0, stream>>>(hxh, 384, whz, gz_b, zb, nullptr, 0, 0, 128, H, W, 2);
    k_wmma_conv<3><<<dim3(HWl / 32, 8), wv, 0, stream>>>(hxh, 384, whr, gr_b, rb, nullptr, 0, 0, 128, H, W, 2);
    k_rnx<<<cdiv(128 * HWl, 256), 256, 0, stream>>>(hxh, rb, net, HWl);      // hx[:,0:128] = r*net
    k_wmma_conv<3><<<dim3(HWl / 32, 8), wv, 0, stream>>>(hxh, 384, whq, gq_b, rb, nullptr, 0, 0, 128, H, W, 3);
    k_net_update<<<cdiv(128 * HWl, 256), 256, 0, stream>>>(net, zb, rb, HWl);
    k_conv3x3_direct<<<cdiv(HWl, 256), 256, 0, stream>>>(net, head_w, head_b, ddisp, nullptr, 0, 0, 128, 1, H, W, 0);
    k_add<<<cdiv(HWl, 256), 256, 0, stream>>>(disp, ddisp, HWl);
  }

  // --- convex upsampling ---
  k_to_f16_pm<<<cdiv(HWl * 128, 256), 256, 0, stream>>>(net, neth, 128, 128, HWl);
  k_wmma_conv<3><<<dim3(HWl / 32, 2), wv, 0, stream>>>(neth, 128, whmk, mask_b, maskf, nullptr, 0, 0, 32, H, W, 1);
  k_cat2x<<<cdiv(64 * 4 * HWl, 256), 256, 0, stream>>>(maskf, stem2x, cat2xh, H, W);
  k_wmma_conv<3><<<dim3(4 * HWl / 32, 2), wv, 0, stream>>>(cat2xh, 64, whs2, spx2_b, xspx, nullptr, 0, 0, 32, 2 * H, 2 * W, 1);
  k_upsample<<<cdiv(16 * HWl, 256), 256, 0, stream>>>(xspx, spxg_w, spxg_b, disp, (float*)d_out, H, W);
}
